// GATLayer_53420803228013
// MI455X (gfx1250) — matrix operand order repaired and checked by reading, not compiled, not run on hardware
//
#include <hip/hip_runtime.h>
#include <hip/hip_bf16.h>
#include <stdint.h>

// GAT layer, MI455X (gfx1250, wave32).
//   B=4, N=4096, Fin=256, Fout=128
// Bound by adj traffic (2 x 256 MB @ 23.3 TB/s ~ 22us); dense math via
// v_wmma_f32_16x16x32_bf16 (~18 GFLOP, a few us). Softmax kept in fp32.
// bf16 conversion via native fptrunc (v_cvt_*_bf16_f32) instead of bit math.

#define GAT_ALPHA 0.2f
#define GAT_NEG_INF (-9.0e15f)

#define B_  4
#define N_  4096
#define FIN 256
#define FOUT 128

typedef __attribute__((ext_vector_type(16))) __bf16        v16bf;
typedef __attribute__((ext_vector_type(16))) unsigned short v16u;
typedef __attribute__((ext_vector_type(8)))  float          v8f;

__device__ __forceinline__ unsigned short f2bfu(float f) {
  __bf16 b = (__bf16)f;  // native v_cvt bf16 (RNE)
  return __builtin_bit_cast(unsigned short, b);
}
__device__ __forceinline__ float bf2f(unsigned short s) {
  return __uint_as_float(((unsigned)s) << 16);
}

// ---------------------------------------------------------------------------
// K0: WT[i*FIN + k] = bf16(W[k*FOUT + i])   (B-operand of GEMM1, contiguous K)
// ---------------------------------------------------------------------------
__global__ __launch_bounds__(256) void k0_transpose_W(
    const float* __restrict__ W, unsigned short* __restrict__ WT) {
  int idx = blockIdx.x * blockDim.x + threadIdx.x;  // 0 .. FOUT*FIN-1
  int i = idx >> 8;        // FIN = 256
  int k = idx & 255;
  WT[idx] = f2bfu(W[k * FOUT + i]);
}

// ---------------------------------------------------------------------------
// K1: Wh = h @ W via WMMA bf16; store transposed bf16: WhT[b][i*N + n]
// block = 256 thr (8 waves); block -> 16 node rows; wave w -> i-cols [16w,16w+16)
// ---------------------------------------------------------------------------
__global__ __launch_bounds__(256) void k1_wh_gemm(
    const float* __restrict__ h, const unsigned short* __restrict__ WT,
    unsigned short* __restrict__ WhT) {
  const int b    = blockIdx.y;
  const int n0   = blockIdx.x * 16;
  const int wave = threadIdx.x >> 5;
  const int lane = threadIdx.x & 31;
  const int lm   = lane & 15;
  const int half = lane >> 4;
  const int i0   = wave * 16;

  const float* hb = h + ((size_t)b * N_ + n0) * FIN;

  v8f c = {};
  for (int kf0 = 0; kf0 < FIN; kf0 += 32) {
    // A tile 16x32 bf16: lane lm = row; halves carry K {off..off+15}, off=16*half
    const int off = half * 16;
    const float* hr = hb + (size_t)lm * FIN + kf0 + off;
    v16bf A;
#pragma unroll
    for (int e = 0; e < 8; ++e) A[e] = (__bf16)hr[e];
#pragma unroll
    for (int e = 0; e < 8; ++e) A[8 + e] = (__bf16)hr[8 + e];

    // B tile 32x16 bf16: lane col i = i0+lm; lanes 0-15 K=kf0..+15, 16-31 +16
    const v16u* bp =
        (const v16u*)(WT + (size_t)(i0 + lm) * FIN + kf0 + half * 16);
    v16bf Bv = __builtin_bit_cast(v16bf, *bp);

    c = __builtin_amdgcn_wmma_f32_16x16x32_bf16(false, A, false, Bv,
                                                (short)0, c, false, false);
  }

  // store transposed: WhT[b][(i0+lm)*N + n]; r=0..7 contiguous -> b128 store
  unsigned short* wt = WhT + (size_t)b * FOUT * N_ + (size_t)(i0 + lm) * N_;
#pragma unroll
  for (int r = 0; r < 8; ++r) {
    int n = n0 + r + 8 * half;  // C layout: lanes0-15 M=r, lanes16-31 M=8+r
    wt[n] = f2bfu(c[r]);
  }
}

// ---------------------------------------------------------------------------
// K2: s1[b,n] = sum_i Wh[n,i]*a1[i]; s2 likewise. Coalesced over n.
// ---------------------------------------------------------------------------
__global__ __launch_bounds__(256) void k2_scores(
    const unsigned short* __restrict__ WhT, const float* __restrict__ a,
    float* __restrict__ s1ws, float* __restrict__ s2ws) {
  int t = blockIdx.x * blockDim.x + threadIdx.x;  // 0 .. B*N-1
  int b = t >> 12;
  int n = t & (N_ - 1);
  const unsigned short* wt = WhT + (size_t)b * FOUT * N_;
  float s1 = 0.f, s2 = 0.f;
#pragma unroll 4
  for (int i = 0; i < FOUT; ++i) {
    float w = bf2f(wt[(size_t)i * N_ + n]);
    s1 = fmaf(w, a[i], s1);
    s2 = fmaf(w, a[FOUT + i], s2);
  }
  s1ws[t] = s1;
  s2ws[t] = s2;
}

// ---------------------------------------------------------------------------
// K3: per-row softmax stats. One wave per (b,j). int4/float4 (b128) streaming
// of the adj row, online (m,l) merge, butterfly reduce; store
// sq[b,j] = {s1[j], q = m + ln(l)} so att = exp(e - q).
// NEG_INF is finite, so empty rows fall out correctly (uniform 1/N).
// ---------------------------------------------------------------------------
__device__ __forceinline__ void online_acc(float e, float& m, float& l) {
  if (e > m) {
    l = l * __expf(m - e) + 1.f;
    m = e;
  } else {
    l += __expf(e - m);
  }
}

__global__ __launch_bounds__(256) void k3_stats(
    const int* __restrict__ adj, const float* __restrict__ s1ws,
    const float* __restrict__ s2ws, float2* __restrict__ sqws) {
  int wid  = (blockIdx.x * blockDim.x + threadIdx.x) >> 5;  // 0 .. B*N-1
  int lane = threadIdx.x & 31;
  int b = wid >> 12;
  int j = wid & (N_ - 1);
  const int* adjr = adj + (size_t)b * N_ * N_ + (size_t)j * N_;
  const float* s2b = s2ws + ((size_t)b << 12);
  float s1j = s1ws[wid];

  float m = -3.0e38f, l = 0.f;
  for (int it = 0; it < N_ / 128; ++it) {  // 128 = 32 lanes * 4 elems
    int k = it * 128 + lane * 4;
    int4   av = *(const int4*)(adjr + k);
    float4 sv = *(const float4*)(s2b + k);
    float x0 = s1j + sv.x, x1 = s1j + sv.y, x2 = s1j + sv.z, x3 = s1j + sv.w;
    x0 = x0 > 0.f ? x0 : GAT_ALPHA * x0;
    x1 = x1 > 0.f ? x1 : GAT_ALPHA * x1;
    x2 = x2 > 0.f ? x2 : GAT_ALPHA * x2;
    x3 = x3 > 0.f ? x3 : GAT_ALPHA * x3;
    online_acc(av.x > 0 ? x0 : GAT_NEG_INF, m, l);
    online_acc(av.y > 0 ? x1 : GAT_NEG_INF, m, l);
    online_acc(av.z > 0 ? x2 : GAT_NEG_INF, m, l);
    online_acc(av.w > 0 ? x3 : GAT_NEG_INF, m, l);
  }
#pragma unroll
  for (int o = 16; o > 0; o >>= 1) {
    float mo = __shfl_xor(m, o);
    float lo = __shfl_xor(l, o);
    float mm = fmaxf(m, mo);
    l = l * __expf(m - mm) + lo * __expf(mo - mm);
    m = mm;
  }
  if (lane == 0) sqws[wid] = make_float2(s1j, m + __logf(l));
}

// ---------------------------------------------------------------------------
// K4: h'[b,k,i] = sum_j att[b,j,k] * Wh[b,j,i], then ELU.
// Workgroup: 128(k) x 128(i) output tile; wave w owns rows [16w,16w+16) x all i.
// Per K-step (32 j): A(16x32 bf16) built in-register from adj + (s1,q,s2);
// 8 WMMAs vs contiguous B tiles from WhT. Each adj element read exactly once.
// ---------------------------------------------------------------------------
__device__ __forceinline__ float att_f32(const int* __restrict__ adjb,
                                         const float2* __restrict__ sqb,
                                         float s2k, int kk, int j) {
  int av = adjb[(size_t)j * N_ + kk];          // coalesced along kk
  float2 sv = sqb[j];                          // {s1[j], m[j]+ln l[j]}
  float x = sv.x + s2k;
  x = x > 0.f ? x : GAT_ALPHA * x;
  float e = av > 0 ? x : GAT_NEG_INF;
  return __expf(e - sv.y);
}

__global__ __launch_bounds__(256) void k4_attn_gemm(
    const int* __restrict__ adj, const unsigned short* __restrict__ WhT,
    const float2* __restrict__ sq, const float* __restrict__ s2ws,
    float* __restrict__ out) {
  const int b    = blockIdx.y;
  const int k0   = blockIdx.x * 128;
  const int wave = threadIdx.x >> 5;
  const int lane = threadIdx.x & 31;
  const int lm   = lane & 15;
  const int half = lane >> 4;

  const int kk = k0 + wave * 16 + lm;  // A-matrix row for this lane
  const int* adjb = adj + (size_t)b * N_ * N_;
  const float2* sqb = sq + ((size_t)b << 12);
  const unsigned short* wt = WhT + (size_t)b * FOUT * N_;
  const float s2k = s2ws[((size_t)b << 12) + kk];

  v8f c[8];
#pragma unroll
  for (int n = 0; n < 8; ++n) c[n] = (v8f){};

  for (int j0 = 0; j0 < N_; j0 += 32) {
    if (j0 + 32 < N_)  // hint next adj tile (global_prefetch)
      __builtin_prefetch(adjb + (size_t)(j0 + 32) * N_ + kk, 0, 1);

    // Build A tile (attn^T, rows=k, K=j) in contiguous K order: element s holds j = j0 + 16*half + s.
    const int off = half * 16;
    v16bf A;
#pragma unroll
    for (int e = 0; e < 8; ++e) {
      A[e]     = (__bf16)att_f32(adjb, sqb, s2k, kk, j0 + off + e);
      A[8 + e] = (__bf16)att_f32(adjb, sqb, s2k, kk, j0 + off + 8 + e);
    }

    const int jb = j0 + half * 16;
#pragma unroll
    for (int n = 0; n < 8; ++n) {
      const v16u* bp = (const v16u*)(wt + (size_t)(n * 16 + lm) * N_ + jb);
      v16bf Bv = __builtin_bit_cast(v16bf, *bp);
      c[n] = __builtin_amdgcn_wmma_f32_16x16x32_bf16(false, A, false, Bv,
                                                     (short)0, c[n], false,
                                                     false);
    }
  }

  // Epilogue: ELU + store fp32
  float* ob = out + (size_t)b * N_ * FOUT;
#pragma unroll
  for (int n = 0; n < 8; ++n) {
#pragma unroll
    for (int r = 0; r < 8; ++r) {
      int row = k0 + wave * 16 + r + 8 * half;
      int col = n * 16 + lm;
      float x = c[n][r];
      x = x > 0.f ? x : (__expf(x) - 1.f);
      ob[(size_t)row * FOUT + col] = x;
    }
  }
}

// ---------------------------------------------------------------------------
extern "C" void kernel_launch(void* const* d_in, const int* in_sizes, int n_in,
                              void* d_out, int out_size, void* d_ws,
                              size_t ws_size, hipStream_t stream) {
  (void)in_sizes; (void)n_in; (void)out_size; (void)ws_size;
  const float* h   = (const float*)d_in[0];
  const int*   adj = (const int*)d_in[1];
  const float* W   = (const float*)d_in[2];
  const float* a   = (const float*)d_in[3];
  float* out = (float*)d_out;

  char* ws = (char*)d_ws;
  unsigned short* WT   = (unsigned short*)(ws);                       // 64 KB
  unsigned short* WhT  = (unsigned short*)(ws + (1 << 16));           // 4 MB
  char* p = ws + (1 << 16) + (size_t)B_ * FOUT * N_ * 2;
  float*  s1ws = (float*)p;  p += (size_t)B_ * N_ * 4;                // 64 KB
  float*  s2ws = (float*)p;  p += (size_t)B_ * N_ * 4;                // 64 KB
  float2* sqws = (float2*)p;                                          // 128 KB

  k0_transpose_W<<<(FOUT * FIN) / 256, 256, 0, stream>>>(W, WT);
  k1_wh_gemm<<<dim3(N_ / 16, B_), 256, 0, stream>>>(h, WT, WhT);
  k2_scores<<<(B_ * N_) / 256, 256, 0, stream>>>(WhT, a, s1ws, s2ws);
  k3_stats<<<(B_ * N_ * 32) / 256, 256, 0, stream>>>(adj, s1ws, s2ws, sqws);
  k4_attn_gemm<<<dim3(N_ / 128, B_), 256, 0, stream>>>(adj, WhT, sqws, s2ws,
                                                       out);
}
